// Attention_3856880632239
// MI455X (gfx1250) — matrix operand order repaired and checked by reading, not compiled, not run on hardware
//
#include <hip/hip_runtime.h>
#include <math.h>

// ---- problem constants (match reference) ----
constexpr int B_   = 4;
constexpr int F_   = 16;
constexpr int NP_  = 196;
constexpr int DIM_ = 512;
constexpr int H_   = 8;
constexpr int DH_  = 64;
constexpr int N_   = 1 + F_ * NP_;   // 3137
constexpr int BH_  = B_ * H_;        // 32
constexpr int M_   = B_ * N_;        // 12548 rows (flattened batch x tokens)
constexpr int NQKV_ = 3 * H_ * DH_;  // 1536

typedef __attribute__((ext_vector_type(16))) _Float16 v16h;
typedef __attribute__((ext_vector_type(8)))  _Float16 v8h;
typedef __attribute__((ext_vector_type(8)))  float    v8f;

// D = A(16x32 f16) * B(32x16 f16) + C(16x16 f32)
__device__ inline v8f wmma_f16(v16h a, v16h b, v8f c) {
  return __builtin_amdgcn_wmma_f32_16x16x32_f16(
      /*neg_a=*/false, a, /*neg_b=*/false, b,
      /*c_mod=*/(short)0, c, /*reuse_a=*/false, /*reuse_b=*/false);
}

// A fragment (16x32, K contiguous per lane-half): lane<16 -> row, K k0..k0+15;
// lane>=16 -> row, K k0+16..k0+31. Two contiguous 16B loads.
__device__ inline v16h load_fragA(const _Float16* base, int ld, int row, int k0, int lane) {
  const _Float16* p = base + (size_t)row * ld + k0 + ((lane >> 4) << 4);
  v8h lo = *(const v8h*)p;
  v8h hi = *(const v8h*)(p + 8);
  v16h r;
#pragma unroll
  for (int i = 0; i < 8; ++i) { r[i] = lo[i]; r[i + 8] = hi[i]; }
  return r;
}

// B fragment (32x16, K contiguous per lane-half): lane<16 -> col, K k0..k0+15;
// lane>=16 -> col, K k0+16..k0+31. One contiguous 32B run.
__device__ inline v16h load_fragB(const _Float16* base, int ld, int col_row, int k0, int lane) {
  const _Float16* p = base + (size_t)col_row * ld + k0 + ((lane >> 4) << 4);
  v8h lo = *(const v8h*)p;
  v8h hi = *(const v8h*)(p + 8);
  v16h r;
#pragma unroll
  for (int i = 0; i < 8; ++i) { r[i] = lo[i]; r[i + 8] = hi[i]; }
  return r;
}

// ---------------- conversion / transpose kernels ----------------
__global__ void cvt_f32_f16_kernel(const float* __restrict__ in, _Float16* __restrict__ out, int n) {
  int i = blockIdx.x * 256 + threadIdx.x;
  if (i < n) out[i] = (_Float16)in[i];
}

// in: rows x cols (row-major f32)  ->  out: cols x rows (row-major f16)
__global__ void transpose_cvt_kernel(const float* __restrict__ in, _Float16* __restrict__ out,
                                     int rows, int cols) {
  int i = blockIdx.x * 256 + threadIdx.x;
  if (i < rows * cols) {
    int r = i / cols, c = i % cols;
    out[(size_t)c * rows + r] = (_Float16)in[i];
  }
}

// ---------------- GEMM 1: qkv = x @ W_qkv, 32x64 tile/wave, scatter into q/k/v ----------------
__global__ __launch_bounds__(32) void gemm_qkv_kernel(const _Float16* __restrict__ xh,
                                                      const _Float16* __restrict__ wT, // (1536 x 512)
                                                      _Float16* __restrict__ q,
                                                      _Float16* __restrict__ k,
                                                      _Float16* __restrict__ v) {
  int lane = threadIdx.x;
  int nb = blockIdx.x * 64;
  int mb = blockIdx.y * 32;
  int ar0 = mb + (lane & 15);       if (ar0 >= M_) ar0 = M_ - 1;
  int ar1 = mb + 16 + (lane & 15);  if (ar1 >= M_) ar1 = M_ - 1;
  int br  = nb + (lane & 15);
  v8f acc[2][4] = {};
#pragma unroll 2
  for (int kk = 0; kk < DIM_; kk += 32) {
    v16h a0 = load_fragA(xh, DIM_, ar0, kk, lane);
    v16h a1 = load_fragA(xh, DIM_, ar1, kk, lane);
    v16h b0 = load_fragB(wT, DIM_, br,      kk, lane);
    v16h b1 = load_fragB(wT, DIM_, br + 16, kk, lane);
    v16h b2 = load_fragB(wT, DIM_, br + 32, kk, lane);
    v16h b3 = load_fragB(wT, DIM_, br + 48, kk, lane);
    acc[0][0] = wmma_f16(a0, b0, acc[0][0]);
    acc[0][1] = wmma_f16(a0, b1, acc[0][1]);
    acc[0][2] = wmma_f16(a0, b2, acc[0][2]);
    acc[0][3] = wmma_f16(a0, b3, acc[0][3]);
    acc[1][0] = wmma_f16(a1, b0, acc[1][0]);
    acc[1][1] = wmma_f16(a1, b1, acc[1][1]);
    acc[1][2] = wmma_f16(a1, b2, acc[1][2]);
    acc[1][3] = wmma_f16(a1, b3, acc[1][3]);
  }
  int rb = (lane >> 4) << 3;
#pragma unroll
  for (int j = 0; j < 4; ++j) {
    int col = nb + j * 16 + (lane & 15);
    int which = col / (H_ * DH_);
    int hh = (col % (H_ * DH_)) / DH_;
    int dd = col % DH_;
#pragma unroll
    for (int i2 = 0; i2 < 2; ++i2) {
#pragma unroll
      for (int i = 0; i < 8; ++i) {
        int m = mb + i2 * 16 + i + rb;
        if (m < M_) {
          int bb = m / N_, n = m % N_;
          size_t idx = (((size_t)(bb * H_ + hh)) * N_ + n) * DH_ + dd;
          float val = acc[i2][j][i];
          if (which == 0)      q[idx] = (_Float16)(val * 0.125f);  // 1/sqrt(64)
          else if (which == 1) k[idx] = (_Float16)val;
          else                 v[idx] = (_Float16)val;
        }
      }
    }
  }
}

// ---------------- CLS attention: per (b,h), 1 query over all 3137 keys ----------------
__global__ __launch_bounds__(256) void cls_attn_kernel(const _Float16* __restrict__ q,
                                                       const _Float16* __restrict__ k,
                                                       const _Float16* __restrict__ v,
                                                       _Float16* __restrict__ ao) {
  __shared__ float sc[N_];
  __shared__ float red[256];
  __shared__ float qs[DH_];
  int bh = blockIdx.x;
  int t  = threadIdx.x;
  const _Float16* qb = q + (size_t)bh * N_ * DH_;
  const _Float16* kb = k + (size_t)bh * N_ * DH_;
  const _Float16* vb = v + (size_t)bh * N_ * DH_;
  if (t < DH_) qs[t] = (float)qb[t];
  __syncthreads();
  for (int j = t; j < N_; j += 256) {
    const _Float16* kp = kb + (size_t)j * DH_;
    float s = 0.f;
#pragma unroll
    for (int d = 0; d < DH_; ++d) s += qs[d] * (float)kp[d];
    sc[j] = s;
  }
  __syncthreads();
  float mx = -3.0e38f;
  for (int j = t; j < N_; j += 256) mx = fmaxf(mx, sc[j]);
  red[t] = mx; __syncthreads();
  for (int s = 128; s > 0; s >>= 1) { if (t < s) red[t] = fmaxf(red[t], red[t + s]); __syncthreads(); }
  mx = red[0]; __syncthreads();
  float sm = 0.f;
  for (int j = t; j < N_; j += 256) { float e = __expf(sc[j] - mx); sc[j] = e; sm += e; }
  red[t] = sm; __syncthreads();
  for (int s = 128; s > 0; s >>= 1) { if (t < s) red[t] += red[t + s]; __syncthreads(); }
  float inv = 1.f / red[0];
  __syncthreads();
  if (t < DH_) {
    float o = 0.f;
    for (int j = 0; j < N_; ++j) o += sc[j] * (float)vb[(size_t)j * DH_ + t];
    int bb = bh / H_, hh = bh % H_;
    ao[(size_t)bb * N_ * (H_ * DH_) + hh * DH_ + t] = (_Float16)(o * inv);
  }
}

// ---------------- Local attention: 512 groups, 196 queries x 197 keys, D=64 ----------------
__global__ __launch_bounds__(32) void local_attn_kernel(const _Float16* __restrict__ q,
                                                        const _Float16* __restrict__ k,
                                                        const _Float16* __restrict__ v,
                                                        _Float16* __restrict__ ao) {
  constexpr int NKEY = NP_ + 1;   // 197
  constexpr int NJT  = 13;        // ceil(197/16)
  constexpr int NKP  = 224;       // padded K for A@V (multiple of 32)
  __shared__ float sim[16][16 * NJT];            // 16 x 208 f32
  __shared__ __align__(16) _Float16 af[16][NKP]; // softmax probs, f16, zero-padded

  int g = blockIdx.x;             // 0..511 = bh*F + f
  int bh = g / F_, fi = g % F_;
  int bb = bh / H_, hh = bh % H_;
  int lane = threadIdx.x;
  const _Float16* qb = q + (size_t)bh * N_ * DH_;
  const _Float16* kb = k + (size_t)bh * N_ * DH_;
  const _Float16* vb = v + (size_t)bh * N_ * DH_;

  for (int mt = 0; mt < 13; ++mt) {
    // ---- S = Q Kt : 16 x 197 scores ----
    int p = mt * 16 + (lane & 15); if (p > NP_ - 1) p = NP_ - 1;
    int qtok = 1 + fi * NP_ + p;
    for (int jt = 0; jt < NJT; ++jt) {
      v8f acc = {};
      int j = jt * 16 + (lane & 15); if (j > NKEY - 1) j = NKEY - 1;
      int ktok = (j == 0) ? 0 : (1 + fi * NP_ + (j - 1));
#pragma unroll
      for (int ks = 0; ks < DH_; ks += 32) {
        v16h a = load_fragA(qb, DH_, qtok, ks, lane);
        v16h b = load_fragB(kb, DH_, ktok, ks, lane);
        acc = wmma_f16(a, b, acc);
      }
      int col = jt * 16 + (lane & 15);
      int rb = (lane >> 4) << 3;
#pragma unroll
      for (int i = 0; i < 8; ++i) sim[i + rb][col] = acc[i];
    }
    __syncthreads();
    // ---- row softmax (fp32), write f16 probs zero-padded to 224 ----
    if (lane < 16) {
      int r = lane;
      float mmax = -3.0e38f;
      for (int j = 0; j < NKEY; ++j) mmax = fmaxf(mmax, sim[r][j]);
      float ssum = 0.f;
      for (int j = 0; j < NKEY; ++j) ssum += __expf(sim[r][j] - mmax);
      float inv = 1.0f / ssum;
      for (int j = 0; j < NKEY; ++j) af[r][j] = (_Float16)(__expf(sim[r][j] - mmax) * inv);
      for (int j = NKEY; j < NKP; ++j) af[r][j] = (_Float16)0.f;
    }
    __syncthreads();
    // ---- O = P V : 16 x 64 ----
    for (int nt = 0; nt < 4; ++nt) {
      v8f acc = {};
      int d = nt * 16 + (lane & 15);
      for (int ks = 0; ks < 7; ++ks) {
        v16h a = load_fragA(&af[0][0], NKP, (lane & 15), ks * 32, lane);
        v16h bf;
        int kb0 = ks * 32 + ((lane >> 4) << 4);
#pragma unroll
        for (int t2 = 0; t2 < 16; ++t2) {
          int j = kb0 + t2; if (j > NKEY - 1) j = NKEY - 1; // padded cols have P==0
          int tok = (j == 0) ? 0 : (1 + fi * NP_ + (j - 1));
          bf[t2] = vb[(size_t)tok * DH_ + d];
        }
        acc = wmma_f16(a, bf, acc);
      }
      int rb = (lane >> 4) << 3;
#pragma unroll
      for (int i = 0; i < 8; ++i) {
        int row = i + rb;
        int pos = mt * 16 + row;
        if (pos < NP_) {
          int tok = 1 + fi * NP_ + pos;
          ao[((size_t)bb * N_ + tok) * (H_ * DH_) + hh * DH_ + d] = (_Float16)acc[i];
        }
      }
    }
    __syncthreads();
  }
}

// ---------------- GEMM 2: out = ao @ W_out + b_out, 32x64 tile/wave (fp32 result) ----------------
__global__ __launch_bounds__(32) void gemm_out_kernel(const _Float16* __restrict__ ah,
                                                      const _Float16* __restrict__ wT, // (512 x 512)
                                                      const float* __restrict__ bias,
                                                      float* __restrict__ out) {
  int lane = threadIdx.x;
  int nb = blockIdx.x * 64;
  int mb = blockIdx.y * 32;
  int ar0 = mb + (lane & 15);       if (ar0 >= M_) ar0 = M_ - 1;
  int ar1 = mb + 16 + (lane & 15);  if (ar1 >= M_) ar1 = M_ - 1;
  int br  = nb + (lane & 15);
  v8f acc[2][4] = {};
#pragma unroll 2
  for (int kk = 0; kk < DIM_; kk += 32) {
    v16h a0 = load_fragA(ah, DIM_, ar0, kk, lane);
    v16h a1 = load_fragA(ah, DIM_, ar1, kk, lane);
    v16h b0 = load_fragB(wT, DIM_, br,      kk, lane);
    v16h b1 = load_fragB(wT, DIM_, br + 16, kk, lane);
    v16h b2 = load_fragB(wT, DIM_, br + 32, kk, lane);
    v16h b3 = load_fragB(wT, DIM_, br + 48, kk, lane);
    acc[0][0] = wmma_f16(a0, b0, acc[0][0]);
    acc[0][1] = wmma_f16(a0, b1, acc[0][1]);
    acc[0][2] = wmma_f16(a0, b2, acc[0][2]);
    acc[0][3] = wmma_f16(a0, b3, acc[0][3]);
    acc[1][0] = wmma_f16(a1, b0, acc[1][0]);
    acc[1][1] = wmma_f16(a1, b1, acc[1][1]);
    acc[1][2] = wmma_f16(a1, b2, acc[1][2]);
    acc[1][3] = wmma_f16(a1, b3, acc[1][3]);
  }
  int rb = (lane >> 4) << 3;
#pragma unroll
  for (int j = 0; j < 4; ++j) {
    int col = nb + j * 16 + (lane & 15);
    float bv = bias[col];
#pragma unroll
    for (int i2 = 0; i2 < 2; ++i2) {
#pragma unroll
      for (int i = 0; i < 8; ++i) {
        int m = mb + i2 * 16 + i + rb;
        if (m < M_) out[(size_t)m * DIM_ + col] = acc[i2][j][i] + bv;
      }
    }
  }
}

extern "C" void kernel_launch(void* const* d_in, const int* in_sizes, int n_in,
                              void* d_out, int out_size, void* d_ws, size_t ws_size,
                              hipStream_t stream) {
  (void)in_sizes; (void)n_in; (void)out_size; (void)ws_size;
  const float* x    = (const float*)d_in[0];
  const float* Wqkv = (const float*)d_in[1];
  const float* Wout = (const float*)d_in[2];
  const float* bout = (const float*)d_in[3];
  // d_in[4] = f (compile-time constant F_ = 16)

  char* ws = (char*)d_ws;
  size_t off = 0;
  auto carve = [&](size_t bytes) -> char* {
    char* p = ws + off;
    off += (bytes + 255) & ~(size_t)255;
    return p;
  };
  _Float16* xh  = (_Float16*)carve((size_t)M_ * DIM_ * sizeof(_Float16));
  _Float16* WqT = (_Float16*)carve((size_t)NQKV_ * DIM_ * sizeof(_Float16));
  _Float16* WoT = (_Float16*)carve((size_t)DIM_ * DIM_ * sizeof(_Float16));
  _Float16* q   = (_Float16*)carve((size_t)BH_ * N_ * DH_ * sizeof(_Float16));
  _Float16* k   = (_Float16*)carve((size_t)BH_ * N_ * DH_ * sizeof(_Float16));
  _Float16* v   = (_Float16*)carve((size_t)BH_ * N_ * DH_ * sizeof(_Float16));
  _Float16* ao  = (_Float16*)carve((size_t)M_ * DIM_ * sizeof(_Float16));

  cvt_f32_f16_kernel<<<(M_ * DIM_ + 255) / 256, 256, 0, stream>>>(x, xh, M_ * DIM_);
  transpose_cvt_kernel<<<(DIM_ * NQKV_ + 255) / 256, 256, 0, stream>>>(Wqkv, WqT, DIM_, NQKV_);
  transpose_cvt_kernel<<<(DIM_ * DIM_ + 255) / 256, 256, 0, stream>>>(Wout, WoT, DIM_, DIM_);

  gemm_qkv_kernel<<<dim3(NQKV_ / 64, (M_ + 31) / 32), 32, 0, stream>>>(xh, WqT, q, k, v);
  cls_attn_kernel<<<BH_, 256, 0, stream>>>(q, k, v, ao);
  local_attn_kernel<<<BH_ * F_, 32, 0, stream>>>(q, k, v, ao);
  gemm_out_kernel<<<dim3(DIM_ / 64, (M_ + 31) / 32), 32, 0, stream>>>(ao, WoT, bout, (float*)d_out);
}
